// Network_31688268710401
// MI455X (gfx1250) — matrix operand order repaired and checked by reading, not compiled, not run on hardware
//
#include <hip/hip_runtime.h>
#include <hip/hip_bf16.h>
#include <math.h>

// ---------------- problem constants ----------------
#define NPTS      131072
#define TSIZE     524288        // 2^19 hash table entries per level
#define NLEV      16
#define PTS_PER_BLOCK 128       // 8 wave32 x 16 points

typedef __attribute__((ext_vector_type(16))) _Float16 v16h;
typedef __attribute__((ext_vector_type(8)))  _Float16 h8;
typedef __attribute__((ext_vector_type(8)))  float    v8f;

// floor(16 * 2^(l/3)) for l=0..15
__constant__ float c_res[NLEV] = {16.f,20.f,25.f,32.f,40.f,50.f,64.f,80.f,
                                  101.f,128.f,161.f,203.f,256.f,322.f,406.f,512.f};

// ---------------- workspace layout ----------------
// [0,64)   : fp_bias_eff, 16 f32  (fp_b + out_b @ fp_w[64:128])
// [64,...) : transposed f16 weights WT[out_feat][K], K padded to mult of 32
enum {
  O_OCC0 = 0,                 // 64 x 32
  O_OCC1 = O_OCC0 + 64*32,    // 64 x 64
  O_OCC2 = O_OCC1 + 64*64,    // 16 x 64
  O_PB   = O_OCC2 + 16*64,    // 64 x 64 (k 60..63 zero)
  O_V    = O_PB   + 64*64,    // 64 x 64
  O_OW   = O_V    + 64*64,    // 64 x 64
  O_FP   = O_OW   + 64*64,    // 16 x 64 (top half of fp_w)
  O_RGB0 = O_FP   + 16*64,    // 64 x 96 (k 90..95 zero)
  O_RGB1 = O_RGB0 + 64*96,    // 64 x 64
  O_RGB2 = O_RGB1 + 64*64,    // 16 x 64 (feat 3..15 zero)
  O_END  = O_RGB2 + 16*64     // 31744 f16 -> ws needs 64 + 63488 bytes
};

// branch-free stable softplus: max(x,0) + log(1 + exp(-|x|)), hardware TRANS ops
__device__ __forceinline__ float softplusf(float x) {
  float e = __expf(-fabsf(x));
  return fmaxf(x, 0.f) + __logf(1.f + e);
}
__device__ __forceinline__ float sigmoidf_(float x) {
  return __frcp_rn(1.f + __expf(-x));
}

__device__ __forceinline__ v16h ldfrag(const _Float16* p0, const _Float16* p1) {
  union { v16h v; h8 h[2]; } u;
  u.h[0] = *(const h8*)p0;
  u.h[1] = *(const h8*)p1;
  return u.v;
}

// GEMM as D = WT_tile x X^T.
//  A (16x32 f16)  lane: M = f0 + (lane&15); halves at K = k0+8*hi (+0..7) and k0+16+8*hi
//  B (32x16 f16)  lane: N(point) = lane&15; halves at K = k0+8*hi (+0..7) and k0+16+8*hi
//  C/D (f32 16x16) lane: point = lane&15, vgpr r -> feature f0 + 8*hi + r
template<int KCH, int FOUT, int ACT>   // ACT: 0=none 1=softplus
__device__ __forceinline__ void gemm_layer(
    const _Float16* __restrict__ sin_, int sInStride,
    const _Float16* __restrict__ wt,  int Kp,
    const float*    __restrict__ bias,
    _Float16* __restrict__ sout, int sOutStride,
    int lrb, int lane)
{
  const int col = lane & 15;
  const int hi  = lane >> 4;
  const int kbA = hi * 8;
  const int kbB = hi * 8;
  const int lr  = lrb + col;
  v16h bfr[KCH];
  #pragma unroll
  for (int kc = 0; kc < KCH; ++kc) {
    const _Float16* rb = sin_ + lr * sInStride + kc * 32 + kbB;
    bfr[kc] = ldfrag(rb, rb + 16);
  }
  #pragma unroll
  for (int f0 = 0; f0 < FOUT; f0 += 16) {
    v8f acc;
    #pragma unroll
    for (int r = 0; r < 8; ++r) acc[r] = bias[f0 + hi * 8 + r];
    #pragma unroll
    for (int kc = 0; kc < KCH; ++kc) {
      const _Float16* ra = wt + (f0 + col) * Kp + kc * 32 + kbA;
      v16h afr = ldfrag(ra, ra + 16);
      acc = __builtin_amdgcn_wmma_f32_16x16x32_f16(
                false, afr, false, bfr[kc], (short)0, acc, false, false);
    }
    union { h8 v; _Float16 s[8]; } o;
    #pragma unroll
    for (int r = 0; r < 8; ++r) {
      float x = acc[r];
      if (ACT == 1) x = softplusf(x);
      o.s[r] = (_Float16)x;
    }
    *(h8*)(sout + lr * sOutStride + f0 + hi * 8) = o.v;
  }
}

// single 16-feature output tile, returns raw f32 accumulator (bias passed in)
template<int KCH>
__device__ __forceinline__ v8f gemm_tile16(
    const _Float16* __restrict__ sin_, int sInStride,
    const _Float16* __restrict__ wt, int Kp,
    int lrb, int lane, v8f acc)
{
  const int col = lane & 15;
  const int hi  = lane >> 4;
  const int kbA = hi * 8;
  const int kbB = hi * 8;
  const int lr  = lrb + col;
  #pragma unroll
  for (int kc = 0; kc < KCH; ++kc) {
    const _Float16* rb = sin_ + lr * sInStride + kc * 32 + kbB;
    v16h bfr = ldfrag(rb, rb + 16);
    const _Float16* ra = wt + col * Kp + kc * 32 + kbA;
    v16h afr = ldfrag(ra, ra + 16);
    acc = __builtin_amdgcn_wmma_f32_16x16x32_f16(
              false, afr, false, bfr, (short)0, acc, false, false);
  }
  return acc;
}

// ---------------- precompute: constant path + weight transpose ----------------
__device__ __forceinline__ void xpose(const float* __restrict__ W, int Kreal, int Nout,
                                      int Kp, int Fout, int Nreal,
                                      _Float16* __restrict__ dst, int tid)
{
  for (int i = tid; i < Fout * Kp; i += 256) {
    int n = i / Kp;
    int k = i - n * Kp;
    float v = (k < Kreal && n < Nreal) ? W[k * Nout + n] : 0.f;
    dst[i] = (_Float16)v;
  }
}

__global__ __launch_bounds__(256) void nerf_precompute(
    const int* __restrict__ latent_index, const float* __restrict__ rgb_latent,
    const float* __restrict__ pa_w, const float* __restrict__ pa_b,
    const float* __restrict__ v_w,  const float* __restrict__ v_b,
    const float* __restrict__ o_w,  const float* __restrict__ o_b,
    const float* __restrict__ fp_w, const float* __restrict__ fp_b,
    const float* __restrict__ occ_w0, const float* __restrict__ occ_w1,
    const float* __restrict__ occ_w2, const float* __restrict__ pb_w,
    const float* __restrict__ rgb_w0, const float* __restrict__ rgb_w1,
    const float* __restrict__ rgb_w2,
    void* __restrict__ ws)
{
  __shared__ float a[64], av[64], ob[64];
  float*     fpb = (float*)ws;
  _Float16*  wtb = (_Float16*)((char*)ws + 64);
  const int tid = threadIdx.x;

  // constant row: a = latent @ pa_w + pa_b ; out_b = ((a@v+v_b)@o+o_b)
  const float* lat = rgb_latent + (size_t)latent_index[0] * 128;
  if (tid < 64) {
    float s = pa_b[tid];
    for (int k = 0; k < 128; ++k) s = fmaf(lat[k], pa_w[k * 64 + tid], s);
    a[tid] = s;
  }
  __syncthreads();
  if (tid < 64) {
    float s = v_b[tid];
    for (int k = 0; k < 64; ++k) s = fmaf(a[k], v_w[k * 64 + tid], s);
    av[tid] = s;
  }
  __syncthreads();
  if (tid < 64) {
    float s = o_b[tid];
    for (int k = 0; k < 64; ++k) s = fmaf(av[k], o_w[k * 64 + tid], s);
    ob[tid] = s;
  }
  __syncthreads();
  if (tid < 16) {  // fold out_b through bottom half of fp_w into the bias
    float s = fp_b[tid];
    for (int k = 0; k < 64; ++k) s = fmaf(ob[k], fp_w[(64 + k) * 16 + tid], s);
    fpb[tid] = s;
  }

  // transpose / convert weights to f16 WT[out][Kpad]
  xpose(occ_w0, 32, 64, 32, 64, 64, wtb + O_OCC0, tid);
  xpose(occ_w1, 64, 64, 64, 64, 64, wtb + O_OCC1, tid);
  xpose(occ_w2, 64, 16, 64, 16, 16, wtb + O_OCC2, tid);
  xpose(pb_w,   60, 64, 64, 64, 64, wtb + O_PB,   tid);
  xpose(v_w,    64, 64, 64, 64, 64, wtb + O_V,    tid);
  xpose(o_w,    64, 64, 64, 64, 64, wtb + O_OW,   tid);
  xpose(fp_w,   64, 16, 64, 16, 16, wtb + O_FP,   tid);  // top 64 rows
  xpose(rgb_w0, 90, 64, 96, 64, 64, wtb + O_RGB0, tid);
  xpose(rgb_w1, 64, 64, 64, 64, 64, wtb + O_RGB1, tid);
  xpose(rgb_w2, 64,  3, 64, 16,  3, wtb + O_RGB2, tid);
}

// ---------------- fused main kernel ----------------
__global__ __launch_bounds__(256) void nerf_fused(
    const float*  __restrict__ tpts, const float* __restrict__ viewdir,
    const float*  __restrict__ feat, const float2* __restrict__ htab,
    const float*  __restrict__ occ_b0, const float* __restrict__ occ_b1,
    const float*  __restrict__ occ_b2, const float* __restrict__ pb_b,
    const float*  __restrict__ v_b,    const float* __restrict__ o_b,
    const float*  __restrict__ rgb_b0, const float* __restrict__ rgb_b1,
    const float*  __restrict__ rgb_b2,
    const void*   __restrict__ ws,
    float* __restrict__ out)
{
  // activation stride 72 f16 (144B: 16B-aligned rows, bank-spread)
  __shared__ _Float16 sXA[PTS_PER_BLOCK * 72];
  __shared__ _Float16 sXB[PTS_PER_BLOCK * 72];
  __shared__ _Float16 sRG[PTS_PER_BLOCK * 104];   // rgb-input rows, 96 used
  __shared__ float    sOcc[PTS_PER_BLOCK];

  const float*    fpb = (const float*)ws;
  const _Float16* wtb = (const _Float16*)((const char*)ws + 64);

  const int tid  = threadIdx.x;
  const int lane = tid & 31;
  const int col  = lane & 15;
  const int hi   = lane >> 4;
  const int lrb  = (tid >> 5) * 16;              // wave-private 16 rows
  const int lr   = lrb + col;                    // block-local row
  const int gp   = blockIdx.x * PTS_PER_BLOCK + lr;   // global point

  // ---- hash encode: this lane's point, levels hi*8 .. hi*8+7 ----
  {
    const float tx = tpts[3 * gp + 0], ty = tpts[3 * gp + 1], tz = tpts[3 * gp + 2];
    #pragma unroll
    for (int li = 0; li < 8; ++li) {
      const int l = hi * 8 + li;
      const float r = c_res[l];
      const float sx = tx * r, sy = ty * r, sz = tz * r;
      const float fx = floorf(sx), fy = floorf(sy), fz = floorf(sz);
      const float dx = sx - fx, dy = sy - fy, dz = sz - fz;
      const unsigned cx = (unsigned)(int)fx, cy = (unsigned)(int)fy, cz = (unsigned)(int)fz;
      float a0 = 0.f, a1 = 0.f;
      #pragma unroll
      for (int ci = 0; ci < 8; ++ci) {
        const unsigned ox = (ci >> 2) & 1, oy = (ci >> 1) & 1, oz = ci & 1;
        unsigned h = (cx + ox) ^ ((cy + oy) * 2654435761u) ^ ((cz + oz) * 805459861u);
        unsigned idx = h & (unsigned)(TSIZE - 1);
        float2 f = htab[(size_t)l * TSIZE + idx];          // L2-resident gather
        float w = (ox ? dx : 1.f - dx) * (oy ? dy : 1.f - dy) * (oz ? dz : 1.f - dz);
        a0 = fmaf(w, f.x, a0);
        a1 = fmaf(w, f.y, a1);
      }
      const _Float16 h0 = (_Float16)a0, h1 = (_Float16)a1;
      sXA[lr * 72 + 2 * l]      = h0;  sXA[lr * 72 + 2 * l + 1]  = h1;
      sRG[lr * 104 + 2 * l]     = h0;  sRG[lr * 104 + 2 * l + 1] = h1;
    }
  }

  // ---- direction frequency encode + zero pad (lanes 0..15 per wave) ----
  if (hi == 0) {
    const float vx = viewdir[3 * gp], vy = viewdir[3 * gp + 1], vz = viewdir[3 * gp + 2];
    _Float16* row = sRG + lr * 104;
    row[32] = (_Float16)vx; row[33] = (_Float16)vy; row[34] = (_Float16)vz;
    #pragma unroll
    for (int j = 0; j < 4; ++j) {
      const float f = (float)(1 << j);
      const int b = 35 + j * 6;
      row[b + 0] = (_Float16)__sinf(vx * f); row[b + 1] = (_Float16)__sinf(vy * f);
      row[b + 2] = (_Float16)__sinf(vz * f); row[b + 3] = (_Float16)__cosf(vx * f);
      row[b + 4] = (_Float16)__cosf(vy * f); row[b + 5] = (_Float16)__cosf(vz * f);
    }
    #pragma unroll
    for (int c = 90; c < 96; ++c) row[c] = (_Float16)0.f;
  }

  // ---- occ MLP: 32 -> 64 -> 64 -> 16 ----
  gemm_layer<1, 64, 1>(sXA, 72, wtb + O_OCC0, 32, occ_b0, sXB, 72, lrb, lane);
  gemm_layer<2, 64, 1>(sXB, 72, wtb + O_OCC1, 64, occ_b1, sXA, 72, lrb, lane);
  {
    v8f acc;
    #pragma unroll
    for (int r = 0; r < 8; ++r) acc[r] = occ_b2[hi * 8 + r];
    acc = gemm_tile16<2>(sXA, 72, wtb + O_OCC2, 64, lrb, lane, acc);
    #pragma unroll
    for (int r = 0; r < 8; ++r) {
      const int f = hi * 8 + r;
      if (f == 0) sOcc[lr] = 1.f - __expf(-softplusf(acc[r]));   // density
      else        sRG[lr * 104 + 58 + f] = (_Float16)acc[r];     // geo feature
    }
  }

  // ---- fuse path: b = feat @ pb_w + pb_b ; out_a = (b@v)@o ; fp ----
  for (int i = lane; i < 16 * 64; i += 32) {
    const int rw = i >> 6, c = i & 63;
    float v = (c < 60) ? feat[(size_t)(blockIdx.x * PTS_PER_BLOCK + lrb + rw) * 60 + c] : 0.f;
    sXB[(lrb + rw) * 72 + c] = (_Float16)v;
  }
  gemm_layer<2, 64, 0>(sXB, 72, wtb + O_PB, 64, pb_b, sXA, 72, lrb, lane);
  gemm_layer<2, 64, 0>(sXA, 72, wtb + O_V,  64, v_b,  sXB, 72, lrb, lane);
  gemm_layer<2, 64, 0>(sXB, 72, wtb + O_OW, 64, o_b,  sXA, 72, lrb, lane);
  {
    v8f acc;
    #pragma unroll
    for (int r = 0; r < 8; ++r) acc[r] = fpb[hi * 8 + r];        // folded bias
    acc = gemm_tile16<2>(sXA, 72, wtb + O_FP, 64, lrb, lane, acc);
    #pragma unroll
    for (int r = 0; r < 8; ++r)
      sRG[lr * 104 + 74 + hi * 8 + r] = (_Float16)acc[r];        // fused -> rgb in
  }

  // ---- rgb MLP: 96 -> 64 -> 64 -> 3 ----
  gemm_layer<3, 64, 1>(sRG, 104, wtb + O_RGB0, 96, rgb_b0, sXB, 72, lrb, lane);
  gemm_layer<2, 64, 1>(sXB, 72,  wtb + O_RGB1, 64, rgb_b1, sXA, 72, lrb, lane);
  {
    v8f acc;
    #pragma unroll
    for (int r = 0; r < 8; ++r) {
      const int f = hi * 8 + r;
      acc[r] = (f < 3) ? rgb_b2[f] : 0.f;
    }
    acc = gemm_tile16<2>(sXA, 72, wtb + O_RGB2, 64, lrb, lane, acc);
    if (hi == 0) {
      float4 o;
      o.x = sigmoidf_(acc[0]);
      o.y = sigmoidf_(acc[1]);
      o.z = sigmoidf_(acc[2]);
      o.w = sOcc[lr];
      *(float4*)(out + (size_t)gp * 4) = o;
    }
  }
}

// ---------------- host entry ----------------
extern "C" void kernel_launch(void* const* d_in, const int* in_sizes, int n_in,
                              void* d_out, int out_size, void* d_ws, size_t ws_size,
                              hipStream_t stream) {
  const float* tpts      = (const float*)d_in[0];
  const float* viewdir   = (const float*)d_in[1];
  // d_in[2] = dists (unused by reference output)
  const float* feat      = (const float*)d_in[3];
  const int*   latent_ix = (const int*)  d_in[4];
  const float2* htab     = (const float2*)d_in[5];
  const float* occ_w0 = (const float*)d_in[6],  *occ_b0 = (const float*)d_in[7];
  const float* occ_w1 = (const float*)d_in[8],  *occ_b1 = (const float*)d_in[9];
  const float* occ_w2 = (const float*)d_in[10], *occ_b2 = (const float*)d_in[11];
  const float* rgb_latent = (const float*)d_in[12];
  const float* pa_w = (const float*)d_in[13], *pa_b = (const float*)d_in[14];
  const float* pb_w = (const float*)d_in[15], *pb_b = (const float*)d_in[16];
  // d_in[17..20] = q_w,q_b,k_w,k_b (unused by reference output)
  const float* v_w  = (const float*)d_in[21], *v_b  = (const float*)d_in[22];
  const float* o_w  = (const float*)d_in[23], *o_b  = (const float*)d_in[24];
  const float* fp_w = (const float*)d_in[25], *fp_b = (const float*)d_in[26];
  const float* rgb_w0 = (const float*)d_in[27], *rgb_b0 = (const float*)d_in[28];
  const float* rgb_w1 = (const float*)d_in[29], *rgb_b1 = (const float*)d_in[30];
  const float* rgb_w2 = (const float*)d_in[31], *rgb_b2 = (const float*)d_in[32];
  float* out = (float*)d_out;

  // ws usage: 64B fp_bias_eff + 63488B f16 transposed weights (= 63552B total)
  nerf_precompute<<<1, 256, 0, stream>>>(
      latent_ix, rgb_latent, pa_w, pa_b, v_w, v_b, o_w, o_b, fp_w, fp_b,
      occ_w0, occ_w1, occ_w2, pb_w, rgb_w0, rgb_w1, rgb_w2, d_ws);

  nerf_fused<<<NPTS / PTS_PER_BLOCK, 256, 0, stream>>>(
      tpts, viewdir, feat, htab,
      occ_b0, occ_b1, occ_b2, pb_b, v_b, o_b,
      rgb_b0, rgb_b1, rgb_b2, d_ws, out);
}
